// Ultra99AccuracyNN_74259984548165
// MI455X (gfx1250) — matrix operand order repaired and checked by reading, not compiled, not run on hardware
//
#include <hip/hip_runtime.h>
#include <hip/hip_bf16.h>
#include <math.h>

// ---------------------------------------------------------------------------
// Fused MLP forward for MI455X (gfx1250, wave32, WMMA 16x16x32 f16).
// BN folded into linears; seq-len-1 attention collapsed to one combined GEMM.
// All intermediates live in LDS; weights pre-converted to f16 [out][in]
// (natural WMMA B-fragment layout) in workspace by small prep kernels.
// Inner loop: all B fragments of a k-step loaded into distinct registers
// (single clause + single wait), then a back-to-back WMMA burst; column
// offset bases hoisted out of the k-loop.
// ---------------------------------------------------------------------------

#define BATCH   131072
#define IN_DIM  43
#define EPS_BN  1e-5f
#define M_TILE  64                  // rows per workgroup
#define THREADS 256                 // 8 wave32
#define SMEM_HALF (2 * 64 * 512 + 64 * 256)   // A + B + R regions (halves)
#define SMEM_BYTES (SMEM_HALF * 2)            // 163840 B of 320 KB LDS

typedef _Float16 h16v __attribute__((ext_vector_type(16)));
typedef _Float16 h8v  __attribute__((ext_vector_type(8)));
typedef float    f8v  __attribute__((ext_vector_type(8)));

enum { MODE_PLAIN = 0, MODE_RELU = 1, MODE_RELU_AFF = 2, MODE_RES = 3 };

// A fragment: 16 contiguous K per lane (matches B), two 16B loads.
__device__ __forceinline__ h16v load_afrag(const _Float16* Arow, int off) {
  h8v alo = *(const h8v*)(Arow + off);        // ds_load_b128
  h8v ahi = *(const h8v*)(Arow + off + 8);    // ds_load_b128
  return __builtin_shufflevector(alo, ahi,
           0,1,2,3,4,5,6,7,8,9,10,11,12,13,14,15);
}

// B fragment: 16 contiguous f16 from W[col][kb]; scalar base + u32 offset so
// the backend emits global_load_b128 saddr+voffset (+imm for unrolled kk).
__device__ __forceinline__ h16v load_bfrag(const _Float16* Bw, unsigned helems) {
  return *(const h16v*)((const char*)Bw + helems * 2u);
}

template<int N, int MODE>
__device__ __forceinline__ void store_tile(
    const f8v& acc, _Float16* __restrict__ Out, const _Float16* __restrict__ Res,
    const float* __restrict__ bias, const float* __restrict__ aff_s,
    const float* __restrict__ aff_t, int rbase, int col)
{
  const float bb = bias[col];
  float s = 0.f, o = 0.f;
  if constexpr (MODE == MODE_RELU_AFF) { s = aff_s[col]; o = aff_t[col]; }
  #pragma unroll
  for (int j = 0; j < 8; ++j) {
    float v = acc[j] + bb;
    if constexpr (MODE == MODE_RELU || MODE == MODE_RELU_AFF) v = fmaxf(v, 0.f);
    if constexpr (MODE == MODE_RELU_AFF) v = v * s + o;
    if constexpr (MODE == MODE_RES) v += (float)Res[(rbase + j) * N + col];
    Out[(rbase + j) * N + col] = (_Float16)v;
  }
}

// ---------------------------------------------------------------------------
// 64xN = (64xK f16 LDS) x (NxK f16 global [out][in]) with fused epilogue.
// ---------------------------------------------------------------------------
template<int K, int N, int MODE>
__device__ __forceinline__ void tile_gemm(
    const _Float16* __restrict__ A,   // LDS [64][K]
    const _Float16* __restrict__ Bw,  // global [N][K] (K multiple of 32)
    const float*    __restrict__ bias,// [N]
    _Float16*       __restrict__ Out, // LDS [64][N]
    const float* __restrict__ aff_s, const float* __restrict__ aff_t,
    const _Float16* __restrict__ Res) // LDS [64][N] (MODE_RES)
{
  const int wave = __builtin_amdgcn_readfirstlane((int)threadIdx.x >> 5); // SGPR
  const int lane = (int)threadIdx.x & 31;
  const int l16  = lane & 15;
  const int lh   = lane >> 4;           // half-wave 0/1
  const int akb  = lh * 16;             // A: 16 contiguous K at lh*16
  const int bkb  = lh * 16;             // B: 16 contiguous K at lh*16
  constexpr int nT = N / 16;
  const f8v zero = {0.f,0.f,0.f,0.f,0.f,0.f,0.f,0.f};

  if constexpr (nT >= 8) {
    constexpr int NW = nT / 8;               // n-tiles per wave
    constexpr int MG = (NW >= 4) ? 2 : 4;    // m-tiles per group (reg budget)

    // per-n-tile column offset bases (k-loop invariant, f16-element units)
    unsigned bbase[NW];
    #pragma unroll
    for (int t = 0; t < NW; ++t)
      bbase[t] = (unsigned)((((wave + 8 * t) << 4) + l16) * K + bkb);

    #pragma unroll
    for (int g = 0; g < 4 / MG; ++g) {
      f8v acc[MG][NW];
      #pragma unroll
      for (int mi = 0; mi < MG; ++mi)
        #pragma unroll
        for (int t = 0; t < NW; ++t) acc[mi][t] = zero;

      const _Float16* Arow[MG];
      #pragma unroll
      for (int mi = 0; mi < MG; ++mi)
        Arow[mi] = A + ((g * MG + mi) * 16 + l16) * K;

      #pragma unroll 2
      for (int kk = 0; kk < K; kk += 32) {
        h16v a[MG];
        h16v b[NW];
        #pragma unroll
        for (int mi = 0; mi < MG; ++mi)          // clause of ds_load_b128
          a[mi] = load_afrag(Arow[mi], kk + akb);
        #pragma unroll
        for (int t = 0; t < NW; ++t)             // clause of global_load_b128
          b[t] = load_bfrag(Bw, bbase[t] + (unsigned)kk);
        #pragma unroll
        for (int t = 0; t < NW; ++t)             // single wait, WMMA burst
          #pragma unroll
          for (int mi = 0; mi < MG; ++mi)
            acc[mi][t] = __builtin_amdgcn_wmma_f32_16x16x32_f16(
                false, a[mi], false, b[t], (short)0, acc[mi][t], false, false);
      }
      #pragma unroll
      for (int mi = 0; mi < MG; ++mi) {
        const int rbase = (g * MG + mi) * 16 + lh * 8;
        #pragma unroll
        for (int t = 0; t < NW; ++t)
          store_tile<N, MODE>(acc[mi][t], Out, Res, bias, aff_s, aff_t,
                              rbase, ((wave + 8 * t) << 4) + l16);
      }
    }
  } else {
    // nT in {1,2,4}: distribute (m,n) pairs across waves, all counts static.
    constexpr int REP = (nT == 4) ? 2 : 1;
    int n, mb;
    if constexpr (nT == 4)      { n = wave & 3; mb = wave >> 2; } // m = mb+2r
    else if constexpr (nT == 2) { n = wave & 1; mb = wave >> 1; } // m = mb
    else                        { n = 0;        mb = wave & 3; }  // dup waves 4-7
    const unsigned bbase = (unsigned)(((n << 4) + l16) * K + bkb);
    #pragma unroll
    for (int r = 0; r < REP; ++r) {
      const int m = (nT == 4) ? (mb + 2 * r) : mb;
      f8v acc = zero;
      const _Float16* Arow = A + (m * 16 + l16) * K;
      #pragma unroll 2
      for (int kk = 0; kk < K; kk += 32) {
        const h16v a = load_afrag(Arow, kk + akb);
        const h16v b = load_bfrag(Bw, bbase + (unsigned)kk);
        acc = __builtin_amdgcn_wmma_f32_16x16x32_f16(
            false, a, false, b, (short)0, acc, false, false);
      }
      store_tile<N, MODE>(acc, Out, Res, bias, aff_s, aff_t,
                          m * 16 + lh * 8, (n << 4) + l16);
    }
  }
}

// ---------------------------------------------------------------------------
struct MP {
  const float* x;  float* out;
  const float *s0, *t0;                         // input BN affine (43)
  const _Float16* W0;  const float* b0;         // 43(->64 pad)->512, BN folded
  const _Float16* C0;  const float* bc0;        // combined attn 512
  const _Float16* rW[4]; const float* rb[4];    // residual linears blk1..4
  const _Float16* Wm[4]; const float* bm[4];    // main linears (BN folded)
  const _Float16* Cm[4]; const float* bcm[4];   // combined attn blk1..4
  const _Float16* Cout; const float* bcout;     // out_attn combined (32)
  const _Float16* H1; const float* hb1; const float *s1, *t1;  // head1+bn1
  const _Float16* H2; const float* hb2; const float *s2, *t2;  // head2+bn2
  const _Float16* H3; const float* hb3;                        // head3
  const float* W4; const float* b4;                            // head4 f32
};

template<int K, int N>
__device__ __forceinline__ void block_step(
    _Float16* A, _Float16* Bb, _Float16* R,
    const _Float16* rW, const float* rb,
    const _Float16* Wm, const float* bm,
    const _Float16* Cm, const float* bcm)
{
  tile_gemm<K, N, MODE_PLAIN>(A, rW, rb, R,  nullptr, nullptr, nullptr);
  tile_gemm<K, N, MODE_RELU >(A, Wm, bm, Bb, nullptr, nullptr, nullptr);
  __syncthreads();
  tile_gemm<N, N, MODE_RES  >(Bb, Cm, bcm, A, nullptr, nullptr, R);
  __syncthreads();
}

__global__ void __launch_bounds__(THREADS) k_fused(MP p) {
  extern __shared__ __attribute__((aligned(16))) _Float16 smem[];
  _Float16* A  = smem;                 // 64 x 512 f16
  _Float16* Bb = smem + 64 * 512;      // 64 x 512 f16
  _Float16* R  = smem + 2 * 64 * 512;  // 64 x 256 f16
  const int r0 = (int)blockIdx.x * M_TILE;

  // Load x tile with input-BN affine, pad 43 -> 64 cols with zeros.
  for (int idx = (int)threadIdx.x; idx < 64 * 64; idx += THREADS) {
    const int row = idx >> 6, col = idx & 63;
    float v = 0.f;
    if (col < IN_DIM)
      v = p.x[(size_t)(r0 + row) * IN_DIM + col] * p.s0[col] + p.t0[col];
    A[idx] = (_Float16)v;
  }
  __syncthreads();

  // block0: main (BN+relu) then combined attn
  tile_gemm<64,  512, MODE_RELU >(A,  p.W0, p.b0,  Bb, nullptr, nullptr, nullptr);
  __syncthreads();
  tile_gemm<512, 512, MODE_PLAIN>(Bb, p.C0, p.bc0, A,  nullptr, nullptr, nullptr);
  __syncthreads();

  // blocks 1..4 (residual)
  block_step<512, 256>(A, Bb, R, p.rW[0], p.rb[0], p.Wm[0], p.bm[0], p.Cm[0], p.bcm[0]);
  block_step<256, 128>(A, Bb, R, p.rW[1], p.rb[1], p.Wm[1], p.bm[1], p.Cm[1], p.bcm[1]);
  block_step<128,  64>(A, Bb, R, p.rW[2], p.rb[2], p.Wm[2], p.bm[2], p.Cm[2], p.bcm[2]);
  block_step< 64,  32>(A, Bb, R, p.rW[3], p.rb[3], p.Wm[3], p.bm[3], p.Cm[3], p.bcm[3]);

  // out_attn (combined)
  tile_gemm<32, 32, MODE_PLAIN>(A, p.Cout, p.bcout, Bb, nullptr, nullptr, nullptr);
  __syncthreads();
  // head: relu+BN-affine, relu+BN-affine, relu
  tile_gemm<32, 64, MODE_RELU_AFF>(Bb, p.H1, p.hb1, A,  p.s1, p.t1, nullptr);
  __syncthreads();
  tile_gemm<64, 32, MODE_RELU_AFF>(A,  p.H2, p.hb2, Bb, p.s2, p.t2, nullptr);
  __syncthreads();
  tile_gemm<32, 16, MODE_RELU>(Bb, p.H3, p.hb3, A, nullptr, nullptr, nullptr);
  __syncthreads();

  // head4 (16 -> 1) + sigmoid, one thread per row
  if (threadIdx.x < M_TILE) {
    const _Float16* ar = A + (size_t)threadIdx.x * 16;
    float acc = p.b4[0];
    #pragma unroll
    for (int k = 0; k < 16; ++k) acc += (float)ar[k] * p.W4[k];
    p.out[r0 + (int)threadIdx.x] = 1.f / (1.f + __expf(-acc));
  }
}

// ---------------------------------------------------------------------------
// Prep kernels (run every launch; total work ~200 MFLOP, fully parallel).
// ---------------------------------------------------------------------------
__global__ void k_bn_fold(const float* bb, const float* g, const float* m,
                          const float* v, float* s, float* t, int n) {
  int i = blockIdx.x * blockDim.x + threadIdx.x;
  if (i < n) {
    float sc = g[i] * rsqrtf(v[i] + EPS_BN);
    s[i] = sc; t[i] = bb[i] - m[i] * sc;
  }
}

// W' = W * bn_scale(row), b' = b*sc + (bn_b - m*sc); pad K -> Kp with zeros
__global__ void k_fold_main(const float* W, const float* b, const float* bnb,
                            const float* g, const float* m, const float* v,
                            _Float16* Wout, float* bout, int K, int Kp, int total) {
  for (int idx = blockIdx.x * blockDim.x + threadIdx.x; idx < total;
       idx += gridDim.x * blockDim.x) {
    int o = idx / Kp, i = idx - o * Kp;
    float sc = g[o] * rsqrtf(v[o] + EPS_BN);
    Wout[idx] = (i < K) ? (_Float16)(W[(size_t)o * K + i] * sc) : (_Float16)0.f;
    if (i == 0) bout[o] = b[o] * sc + (bnb[o] - m[o] * sc);
  }
}

__global__ void k_convert(const float* W, _Float16* Wout, int K, int Kp, int total) {
  for (int idx = blockIdx.x * blockDim.x + threadIdx.x; idx < total;
       idx += gridDim.x * blockDim.x) {
    int o = idx / Kp, i = idx - o * Kp;
    Wout[idx] = (i < K) ? (_Float16)W[(size_t)o * K + i] : (_Float16)0.f;
  }
}

// C = Wo @ Wv (HxH), bc = Wo @ bv + bo  (collapsed seq-len-1 attention)
__global__ void k_combine(const float* Wo, const float* Wv, const float* bo,
                          const float* bv, _Float16* C, float* bc, int H) {
  int total = H * H;
  for (int idx = blockIdx.x * blockDim.x + threadIdx.x; idx < total;
       idx += gridDim.x * blockDim.x) {
    int o = idx / H, i = idx - o * H;
    float acc = 0.f;
    for (int k = 0; k < H; ++k)
      acc += Wo[(size_t)o * H + k] * Wv[(size_t)k * H + i];
    C[idx] = (_Float16)acc;
    if (i == 0) {
      float ba = bo[o];
      for (int k = 0; k < H; ++k) ba += Wo[(size_t)o * H + k] * bv[k];
      bc[o] = ba;
    }
  }
}

// ---------------------------------------------------------------------------
extern "C" void kernel_launch(void* const* d_in, const int* in_sizes, int n_in,
                              void* d_out, int out_size, void* d_ws, size_t ws_size,
                              hipStream_t stream) {
  if (n_in != 85) return;
  auto F = [&](int i) { return (const float*)d_in[i]; };

  // ---- input leaf mapping (sorted-key jax tree order vs insertion order) ---
  const bool insertion = (in_sizes[0] == BATCH * IN_DIM);
  const float *x, *ibn_b, *ibn_g, *ibn_m, *ibn_v;
  const float *bW[5], *bB[5], *bBnB[5], *bBnG[5], *bBnM[5], *bBnV[5];
  const float *bWo[5], *bWv[5], *bBo[5], *bBv[5], *bRW[5], *bRB[5];
  const float *oWo, *oWv, *oBo, *oBv;
  const float *hW1, *hW2, *hW3, *hW4, *hB1, *hB2, *hB3, *hB4;
  const float *h1b, *h1g, *h1m, *h1v, *h2b, *h2g, *h2m, *h2v;

  if (!insertion) {  // jax sorted-dict-key flatten: blocks, head, input_bn, out_attn, x
    for (int i = 0; i < 5; ++i) {
      int b = 12 * i;  // [W, Wo, Wv, b, bn.b, bn.g, bn.m, bn.v, bo, bv, rW, rb]
      bW[i]=F(b); bWo[i]=F(b+1); bWv[i]=F(b+2); bB[i]=F(b+3);
      bBnB[i]=F(b+4); bBnG[i]=F(b+5); bBnM[i]=F(b+6); bBnV[i]=F(b+7);
      bBo[i]=F(b+8); bBv[i]=F(b+9); bRW[i]=F(b+10); bRB[i]=F(b+11);
    }
    hW1=F(60); hW2=F(61); hW3=F(62); hW4=F(63);
    hB1=F(64); hB2=F(65); hB3=F(66); hB4=F(67);
    h1b=F(68); h1g=F(69); h1m=F(70); h1v=F(71);
    h2b=F(72); h2g=F(73); h2m=F(74); h2v=F(75);
    ibn_b=F(76); ibn_g=F(77); ibn_m=F(78); ibn_v=F(79);
    oWo=F(80); oWv=F(81); oBo=F(82); oBv=F(83);
    x = F(84);
  } else {           // python insertion order: x, input_bn, blocks, out_attn, head
    x = F(0);
    ibn_g=F(1); ibn_b=F(2); ibn_m=F(3); ibn_v=F(4);
    for (int i = 0; i < 5; ++i) {
      int b = 5 + 12 * i;  // [W, b, bn.g, bn.b, bn.m, bn.v, rW, rb, Wv, bv, Wo, bo]
      bW[i]=F(b); bB[i]=F(b+1); bBnG[i]=F(b+2); bBnB[i]=F(b+3);
      bBnM[i]=F(b+4); bBnV[i]=F(b+5); bRW[i]=F(b+6); bRB[i]=F(b+7);
      bWv[i]=F(b+8); bBv[i]=F(b+9); bWo[i]=F(b+10); bBo[i]=F(b+11);
    }
    oWv=F(65); oBv=F(66); oWo=F(67); oBo=F(68);
    hW1=F(69); hB1=F(70); h1g=F(71); h1b=F(72); h1m=F(73); h1v=F(74);
    hW2=F(75); hB2=F(76); h2g=F(77); h2b=F(78); h2m=F(79); h2v=F(80);
    hW3=F(81); hB3=F(82); hW4=F(83); hB4=F(84);
  }

  // ---- workspace layout --------------------------------------------------
  char* wsp = (char*)d_ws; size_t off = 0;
  auto a16 = [&](size_t n) {
    _Float16* p = (_Float16*)(wsp + off);
    off = (off + n * 2 + 255) & ~(size_t)255; return p; };
  auto a32 = [&](size_t n) {
    float* p = (float*)(wsp + off);
    off = (off + n * 4 + 255) & ~(size_t)255; return p; };

  const int Kin[5] = {IN_DIM, 512, 256, 128, 64};
  const int Kp[5]  = {64, 512, 256, 128, 64};
  const int Nh[5]  = {512, 256, 128, 64, 32};

  _Float16 *Wp[5], *Cc[5], *rWp[4];
  float *bf[5], *bcc[5];
  for (int i = 0; i < 5; ++i) {
    Wp[i] = a16((size_t)Nh[i] * Kp[i]);  bf[i]  = a32(Nh[i]);
    Cc[i] = a16((size_t)Nh[i] * Nh[i]);  bcc[i] = a32(Nh[i]);
    if (i > 0) rWp[i - 1] = a16((size_t)Nh[i] * Kin[i]);
  }
  _Float16* Cout = a16(32 * 32);  float* bcout = a32(32);
  _Float16* H1p  = a16(64 * 32);
  _Float16* H2p  = a16(32 * 64);
  _Float16* H3p  = a16(16 * 32);
  float *s0 = a32(IN_DIM), *t0 = a32(IN_DIM);
  float *s1 = a32(64), *t1 = a32(64), *s2 = a32(32), *t2 = a32(32);
  if (off > ws_size) return;

  // ---- prep: BN folds, attn combines, f16 conversion ---------------------
  k_bn_fold<<<1, 256, 0, stream>>>(ibn_b, ibn_g, ibn_m, ibn_v, s0, t0, IN_DIM);
  k_bn_fold<<<1, 256, 0, stream>>>(h1b, h1g, h1m, h1v, s1, t1, 64);
  k_bn_fold<<<1, 256, 0, stream>>>(h2b, h2g, h2m, h2v, s2, t2, 32);
  for (int i = 0; i < 5; ++i) {
    int totW = Nh[i] * Kp[i];
    k_fold_main<<<(totW + 255) / 256, 256, 0, stream>>>(
        bW[i], bB[i], bBnB[i], bBnG[i], bBnM[i], bBnV[i],
        Wp[i], bf[i], Kin[i], Kp[i], totW);
    int totC = Nh[i] * Nh[i];
    k_combine<<<(totC + 255) / 256, 256, 0, stream>>>(
        bWo[i], bWv[i], bBo[i], bBv[i], Cc[i], bcc[i], Nh[i]);
    if (i > 0) {
      int totR = Nh[i] * Kin[i];
      k_convert<<<(totR + 255) / 256, 256, 0, stream>>>(
          bRW[i], rWp[i - 1], Kin[i], Kin[i], totR);
    }
  }
  k_combine<<<(32 * 32 + 255) / 256, 256, 0, stream>>>(oWo, oWv, oBo, oBv, Cout, bcout, 32);
  k_convert<<<(64 * 32 + 255) / 256, 256, 0, stream>>>(hW1, H1p, 32, 32, 64 * 32);
  k_convert<<<(32 * 64 + 255) / 256, 256, 0, stream>>>(hW2, H2p, 64, 64, 32 * 64);
  k_convert<<<(16 * 32 + 255) / 256, 256, 0, stream>>>(hW3, H3p, 32, 32, 16 * 32);

  // ---- fused forward -----------------------------------------------------
  MP p;
  p.x = x; p.out = (float*)d_out; p.s0 = s0; p.t0 = t0;
  p.W0 = Wp[0]; p.b0 = bf[0]; p.C0 = Cc[0]; p.bc0 = bcc[0];
  for (int i = 0; i < 4; ++i) {
    p.rW[i] = rWp[i];    p.rb[i]  = bRB[i + 1];
    p.Wm[i] = Wp[i + 1]; p.bm[i]  = bf[i + 1];
    p.Cm[i] = Cc[i + 1]; p.bcm[i] = bcc[i + 1];
  }
  p.Cout = Cout; p.bcout = bcout;
  p.H1 = H1p; p.hb1 = hB1; p.s1 = s1; p.t1 = t1;
  p.H2 = H2p; p.hb2 = hB2; p.s2 = s2; p.t2 = t2;
  p.H3 = H3p; p.hb3 = hB3; p.W4 = hW4; p.b4 = hB4;

  k_fused<<<BATCH / M_TILE, THREADS, SMEM_BYTES, stream>>>(p);
}
